// Extractor_Processor_75368086110414
// MI455X (gfx1250) — compile-verified, matrix operand order checked by reading, not run on hardware
//
#include <hip/hip_runtime.h>
#include <hip/hip_bf16.h>

// ---------------------------------------------------------------------------
// ViTDet block for MI455X (gfx1250): every GEMM-shaped op runs on
// v_wmma_f32_16x16x32_f16 (f16 inputs, f32 accumulate). The generic GEMM's
// A-tile is staged into LDS via the Tensor Data Mover (TENSOR_LOAD_TO_LDS,
// TENSORcnt-tracked) when the builtin is available; otherwise via cooperative
// b128 copies. Weights are converted fp32->f16 once per launch.
// ---------------------------------------------------------------------------

typedef __attribute__((ext_vector_type(16))) _Float16 v16h;
typedef __attribute__((ext_vector_type(8)))  float    v8f;
typedef unsigned int u32x4 __attribute__((ext_vector_type(4)));
typedef int          i32x4 __attribute__((ext_vector_type(4)));
typedef int          i32x8 __attribute__((ext_vector_type(8)));

#define TOKENS 12544   // 16*28*28
#define SN 224         // padded attention K/N stride (196 -> 224, mult of 32)

#if defined(__has_builtin)
#  if __has_builtin(__builtin_amdgcn_tensor_load_to_lds)
#    define HAVE_TDM 1
#  endif
#  if __has_builtin(__builtin_amdgcn_s_wait_tensorcnt)
#    define HAVE_WAIT_TENSOR 1
#  endif
#endif

union Frag { v16h v; uint4 q[2]; };

__device__ inline uint4 zero4(){ uint4 z; z.x = 0u; z.y = 0u; z.z = 0u; z.w = 0u; return z; }

__device__ inline float gelu_f(float x){
  return 0.5f * x * (1.0f + erff(x * 0.70710678118654752f));
}

// spatial row (b*784 + h*28 + w) -> window-ordered row (win*196 + local)
__device__ inline int spatial_to_win(int s){
  int b = s / 784, r = s - b*784;
  int h = r / 28, w = r - (r/28)*28;
  int win = b*4 + (h/14)*2 + (w/14);
  return win*196 + (h%14)*14 + (w%14);
}
// window-ordered row -> spatial row
__device__ inline int win_to_spatial(int r){
  int win = r / 196, t = r - win*196;
  int b = win >> 2, q = win & 3;
  int h = (q >> 1)*14 + t/14;
  int w = (q & 1)*14 + t - (t/14)*14;
  return (b*28 + h)*28 + w;
}

#ifdef HAVE_TDM
// Issue a 2D TDM tile load: `rows` rows of `rowElems` f16 elements, global row
// stride `strideElems`, packed contiguously into LDS at byte offset lds_off.
// D# layout per CDNA5 ISA 8.3-8.6 (count=1, type=2 "image", data_size=1: 2B).
__device__ inline void tdm_load_2d_f16(const void* gptr, unsigned lds_off,
                                       unsigned rowElems, unsigned rows,
                                       unsigned strideElems){
  unsigned long long ga = (unsigned long long)(uintptr_t)gptr;
  u32x4 g0;
  g0[0] = 1u;                                       // count=1 (valid), user mode
  g0[1] = lds_off;                                  // lds_addr (bytes)
  g0[2] = (unsigned)(ga & 0xFFFFFFFFu);             // global_addr[31:0]
  g0[3] = (unsigned)((ga >> 32) & 0x01FFFFFFu)      // global_addr[56:32]
        | (2u << 30);                               // type=2
  i32x8 g1;
  g1[0] = (int)(1u << 16);                          // wg_mask=0, data_size=1 (2B)
  g1[1] = (int)((rowElems & 0xFFFFu) << 16);        // tensor_dim0[15:0] @bit48
  g1[2] = (int)(((rowElems >> 16) & 0xFFFFu)        // tensor_dim0[31:16]
        | ((rows & 0xFFFFu) << 16));                // tensor_dim1[15:0]
  g1[3] = (int)(((rows >> 16) & 0xFFFFu)            // tensor_dim1[31:16]
        | ((rowElems & 0xFFFFu) << 16));            // tile_dim0
  g1[4] = (int)(rows & 0xFFFFu);                    // tile_dim1 (tile_dim2=0)
  g1[5] = (int)strideElems;                         // tensor_dim0_stride[31:0]
  g1[6] = 0;                                        // stride[47:32]=0, d1stride=0
  g1[7] = 0;
  i32x4 gz; gz[0] = 0; gz[1] = 0; gz[2] = 0; gz[3] = 0;
#if __clang_major__ >= 23
  i32x8 gz8; gz8[0]=0; gz8[1]=0; gz8[2]=0; gz8[3]=0; gz8[4]=0; gz8[5]=0; gz8[6]=0; gz8[7]=0;
  __builtin_amdgcn_tensor_load_to_lds(g0, g1, gz, gz, gz8, 0);
#else
  __builtin_amdgcn_tensor_load_to_lds(g0, g1, gz, gz, 0);
#endif
}

__device__ inline void tdm_wait(){
#ifdef HAVE_WAIT_TENSOR
  __builtin_amdgcn_s_wait_tensorcnt(0);
#else
  asm volatile("s_wait_tensorcnt 0x0" ::: "memory");
#endif
}
#endif // HAVE_TDM

// ---------------------------------------------------------------------------
// Elementwise weight conversion / repack
// ---------------------------------------------------------------------------
__global__ __launch_bounds__(256) void cvt_f16(const float* __restrict__ src,
                                               _Float16* __restrict__ dst, int n){
  for (int i = blockIdx.x*256 + threadIdx.x; i < n; i += gridDim.x*256)
    dst[i] = (_Float16)src[i];
}

// conv2_w [co][ci][3][3] fp32 -> [co][p=kh*3+kw][ci] f16 so GEMM-K is contiguous
__global__ __launch_bounds__(256) void repack_conv2(const float* __restrict__ src,
                                                    _Float16* __restrict__ dst){
  const int n = 512*4608;
  for (int i = blockIdx.x*256 + threadIdx.x; i < n; i += gridDim.x*256){
    int co = i / 4608, r = i - co*4608;
    int ci = r / 9,    p = r - ci*9;
    dst[(size_t)co*4608 + p*512 + ci] = (_Float16)src[i];
  }
}

// ---------------------------------------------------------------------------
// Row LayerNorm (last dim D), fp32 in -> f16 out, optional GELU, optional
// window-reorder of the output row (used for LN1 feeding windowed attention).
// ---------------------------------------------------------------------------
__global__ __launch_bounds__(256) void ln_rows(const float* __restrict__ in,
    const float* __restrict__ w, const float* __restrict__ b,
    _Float16* __restrict__ outh, int D, float eps, int winReorder, int doGelu)
{
  __shared__ float red[256];
  const int tid = threadIdx.x;
  const int row = blockIdx.x;
  const float* ip = in + (size_t)row*D;
  float s = 0.f, s2 = 0.f;
  for (int c = tid; c < D; c += 256){ float v = ip[c]; s += v; s2 += v*v; }
  red[tid] = s; __syncthreads();
  for (int st = 128; st > 0; st >>= 1){ if (tid < st) red[tid] += red[tid+st]; __syncthreads(); }
  const float mean = red[0] / (float)D;
  __syncthreads();
  red[tid] = s2; __syncthreads();
  for (int st = 128; st > 0; st >>= 1){ if (tid < st) red[tid] += red[tid+st]; __syncthreads(); }
  const float var  = red[0] / (float)D - mean*mean;
  const float rstd = rsqrtf(var + eps);
  const int orow = winReorder ? spatial_to_win(row) : row;
  _Float16* op = outh + (size_t)orow*D;
  for (int c = tid; c < D; c += 256){
    float v = (ip[c] - mean)*rstd*w[c] + b[c];
    if (doGelu) v = gelu_f(v);
    op[c] = (_Float16)v;
  }
}

// Final: out = res + LN_channel(r3) (eps 1e-6), fp32 out, D=1024
__global__ __launch_bounds__(256) void final_ln_add(const float* __restrict__ r3,
    const float* __restrict__ w, const float* __restrict__ b,
    const float* __restrict__ res, float* __restrict__ out)
{
  __shared__ float red[256];
  const int tid = threadIdx.x;
  const int row = blockIdx.x;
  const float* ip = r3 + (size_t)row*1024;
  float s = 0.f, s2 = 0.f;
  for (int c = tid; c < 1024; c += 256){ float v = ip[c]; s += v; s2 += v*v; }
  red[tid] = s; __syncthreads();
  for (int st = 128; st > 0; st >>= 1){ if (tid < st) red[tid] += red[tid+st]; __syncthreads(); }
  const float mean = red[0] * (1.f/1024.f);
  __syncthreads();
  red[tid] = s2; __syncthreads();
  for (int st = 128; st > 0; st >>= 1){ if (tid < st) red[tid] += red[tid+st]; __syncthreads(); }
  const float var  = red[0] * (1.f/1024.f) - mean*mean;
  const float rstd = rsqrtf(var + 1e-6f);
  for (int c = tid; c < 1024; c += 256)
    out[(size_t)row*1024 + c] = res[(size_t)row*1024 + c]
                              + (ip[c] - mean)*rstd*w[c] + b[c];
}

// ---------------------------------------------------------------------------
// Generic WMMA GEMM:  C[M][N] = A[M][K] (f16) @ W[N][K]^T (f16), f32 accum.
// Block = 256 threads = 8 waves; block tile = 16(M) x 128(N); each wave owns
// one 16x16 C tile. A block-row (16 x K halves) staged in dynamic LDS via the
// Tensor Data Mover (single 2D-tile DMA, TENSORcnt-tracked).
// ---------------------------------------------------------------------------
enum { EPI_BIAS_F16 = 0, EPI_BIAS_GELU_F16 = 1, EPI_BIAS_RES_REMAP_F32 = 2,
       EPI_BIAS_RES_F32_F16 = 3, EPI_F32 = 4 };

template<int EPI>
__global__ __launch_bounds__(256) void gemm_wmma(
    const _Float16* __restrict__ A, const _Float16* __restrict__ Bw,
    const float* __restrict__ bias, const float* __restrict__ res,
    float* __restrict__ outf, _Float16* __restrict__ outh,
    int M, int N, int K)
{
  extern __shared__ __align__(16) _Float16 sA[];   // 16 rows x K halves (LDS off 0)
  const int tid  = threadIdx.x;
  const int lane = tid & 31;
  const int wave = tid >> 5;
  const int m0 = blockIdx.x << 4;
  const int n0 = (blockIdx.y << 7) + (wave << 4);

#ifdef HAVE_TDM
  // Tensor Data Mover: one 2D tile (16 rows x K f16) HBM/L2 -> LDS.
  if (wave == 0){
    tdm_load_2d_f16(A + (size_t)m0*K, 0u, (unsigned)K, 16u, (unsigned)K);
    tdm_wait();
  }
  __syncthreads();
#else
  // fallback: cooperative stage of the A tile (b128 copies)
  const int kch = K >> 3;                // 16B chunks per row
  for (int idx = tid; idx < (kch << 4); idx += 256){
    int row = idx / kch, c = idx - row*kch;
    *(uint4*)(sA + (size_t)row*K + (c<<3)) =
        *(const uint4*)(A + (size_t)(m0 + row)*K + (c<<3));
  }
  __syncthreads();
#endif

  v8f acc = {};
  const _Float16* arow = sA + (size_t)(lane & 15)*K + ((lane>>4)<<3);
  const _Float16* brow = Bw + (size_t)(n0 + (lane & 15))*K + ((lane>>4)<<3);
  for (int k0 = 0; k0 < K; k0 += 32){
    Frag fa, fb;
    fa.q[0] = *(const uint4*)(arow + k0);
    fa.q[1] = *(const uint4*)(arow + k0 + 16);
    fb.q[0] = *(const uint4*)(brow + k0);
    fb.q[1] = *(const uint4*)(brow + k0 + 16);
    __builtin_prefetch(brow + k0 + 64, 0, 0);     // global_prefetch_b8
    acc = __builtin_amdgcn_wmma_f32_16x16x32_f16(false, fa.v, false, fb.v,
                                                 (short)0, acc, false, false);
  }

  const int col   = n0 + (lane & 15);
  const int rbase = m0 + ((lane >> 4) << 3);
  float bv = 0.f;
  if (EPI != EPI_F32) bv = bias[col];
  #pragma unroll
  for (int e = 0; e < 8; ++e){
    int row  = rbase + e;
    float xv = acc[e] + bv;
    if (EPI == EPI_BIAS_F16){
      outh[(size_t)row*N + col] = (_Float16)xv;
    } else if (EPI == EPI_BIAS_GELU_F16){
      outh[(size_t)row*N + col] = (_Float16)gelu_f(xv);
    } else if (EPI == EPI_BIAS_RES_REMAP_F32){
      int sp = win_to_spatial(row);
      outf[(size_t)sp*N + col] = xv + res[(size_t)sp*N + col];
    } else if (EPI == EPI_BIAS_RES_F32_F16){
      float y = xv + res[(size_t)row*N + col];
      outf[(size_t)row*N + col] = y;
      outh[(size_t)row*N + col] = (_Float16)y;
    } else { // EPI_F32
      outf[(size_t)row*N + col] = xv;
    }
  }
}

// ---------------------------------------------------------------------------
// Fused windowed attention per (window, head). Whole 196x196 logit matrix
// (padded to 208x224) lives in LDS as f16; V staged transposed in LDS.
// S = (q*scale)@k^T (WMMA) -> +rel-pos bias -> softmax (LDS) -> S@V (WMMA).
// ---------------------------------------------------------------------------
__global__ __launch_bounds__(256) void attn_win(
    const _Float16* __restrict__ qkv,   // [12544][3072], window-ordered rows
    const float* __restrict__ rph,      // [27][64]
    const float* __restrict__ rpw,      // [27][64]
    _Float16* __restrict__ outh)        // [12544][1024], window-ordered rows
{
  __shared__ __align__(16) _Float16 sS[208*SN];   // logits / softmax (f16)
  __shared__ __align__(16) _Float16 sV[64*SN];    // V transposed: [d][t]
  __shared__ float sRh[196*14];
  __shared__ float sRw[196*14];

  const int hw   = blockIdx.x;          // 0..1023
  const int win  = hw >> 4;
  const int head = hw & 15;
  const int tid  = threadIdx.x;
  const int lane = tid & 31;
  const int wave = tid >> 5;
  const size_t rowbase = (size_t)win * 196;

  // stage V transposed (B-matrix layout [N=d][K=t], K contiguous)
  for (int idx = tid; idx < 64*SN; idx += 256){
    int d = idx / SN, t = idx - d*SN;
    _Float16 val = (_Float16)0.f;
    if (t < 196) val = qkv[(rowbase + t)*3072 + 2048 + head*64 + d];
    sV[idx] = val;
  }
  // decomposed relative-position tables: relh[q][hk], relw[q][wk]
  for (int idx = tid; idx < 196*14; idx += 256){
    int q  = idx / 14, kk = idx - q*14;
    int hq = q / 14,   wq = q - (q/14)*14;
    const _Float16* qp = qkv + (rowbase + q)*3072 + head*64;
    const float* Rh = rph + (hq - kk + 13)*64;
    const float* Rw = rpw + (wq - kk + 13)*64;
    float sh = 0.f, sw = 0.f;
    for (int d = 0; d < 64; ++d){
      float qv = (float)qp[d];
      sh += qv*Rh[d];
      sw += qv*Rw[d];
    }
    sRh[idx] = sh; sRw[idx] = sw;
  }
  __syncthreads();

  // S = q @ k^T (scaled), 13x13 tiles of 16x16, K=64 (2 WMMA steps)
  for (int t = wave; t < 169; t += 8){
    int ti = t / 13, tj = t - ti*13;
    int m0 = ti << 4, n0 = tj << 4;
    int ar = m0 + (lane & 15);
    int br = n0 + (lane & 15);
    v8f acc = {};
    for (int k0 = 0; k0 < 64; k0 += 32){
      Frag fa, fb;
      if (ar < 196){
        const _Float16* p = qkv + (rowbase + ar)*3072 + head*64 + k0 + ((lane>>4)<<3);
        fa.q[0] = *(const uint4*)p; fa.q[1] = *(const uint4*)(p + 16);
      } else { fa.q[0] = zero4(); fa.q[1] = zero4(); }
      if (br < 196){
        const _Float16* p = qkv + (rowbase + br)*3072 + 1024 + head*64 + k0 + ((lane>>4)<<3);
        fb.q[0] = *(const uint4*)p; fb.q[1] = *(const uint4*)(p + 16);
      } else { fb.q[0] = zero4(); fb.q[1] = zero4(); }
      acc = __builtin_amdgcn_wmma_f32_16x16x32_f16(false, fa.v, false, fb.v,
                                                   (short)0, acc, false, false);
    }
    #pragma unroll
    for (int e = 0; e < 8; ++e){
      int r = m0 + e + ((lane>>4)<<3);
      int c = n0 + (lane & 15);
      sS[r*SN + c] = (_Float16)(acc[e] * 0.125f);   // HD^-0.5
    }
  }
  __syncthreads();

  // softmax rows (with rel-pos bias), write normalized f16 back; zero pads
  for (int q = tid; q < 208; q += 256){
    if (q < 196){
      float mx = -3.0e38f;
      for (int n = 0; n < 196; ++n){
        float v = (float)sS[q*SN + n] + sRh[q*14 + n/14] + sRw[q*14 + n%14];
        mx = fmaxf(mx, v);
      }
      float sum = 0.f;
      for (int n = 0; n < 196; ++n){
        float v = __expf((float)sS[q*SN + n] + sRh[q*14 + n/14] + sRw[q*14 + n%14] - mx);
        sum += v;
        sS[q*SN + n] = (_Float16)v;
      }
      float inv = 1.f / sum;
      for (int n = 0; n < 196; ++n) sS[q*SN + n] = (_Float16)((float)sS[q*SN + n]*inv);
      for (int n = 196; n < SN; ++n) sS[q*SN + n] = (_Float16)0.f;
    } else {
      for (int n = 0; n < SN; ++n) sS[q*SN + n] = (_Float16)0.f;
    }
  }
  __syncthreads();

  // out = softmax(S) @ V : tiles 13(M) x 4(N=64), K = 224 (7 WMMA steps)
  for (int t = wave; t < 52; t += 8){
    int ti = t >> 2, tj = t & 3;
    int m0 = ti << 4, n0 = tj << 4;
    const _Float16* pa = sS + (size_t)(m0 + (lane & 15))*SN + ((lane>>4)<<3);
    const _Float16* pb = sV + (size_t)(n0 + (lane & 15))*SN + ((lane>>4)<<3);
    v8f acc = {};
    for (int k0 = 0; k0 < SN; k0 += 32){
      Frag fa, fb;
      fa.q[0] = *(const uint4*)(pa + k0); fa.q[1] = *(const uint4*)(pa + k0 + 16);
      fb.q[0] = *(const uint4*)(pb + k0); fb.q[1] = *(const uint4*)(pb + k0 + 16);
      acc = __builtin_amdgcn_wmma_f32_16x16x32_f16(false, fa.v, false, fb.v,
                                                   (short)0, acc, false, false);
    }
    #pragma unroll
    for (int e = 0; e < 8; ++e){
      int m = m0 + e + ((lane>>4)<<3);
      if (m < 196)
        outh[(rowbase + m)*1024 + head*64 + n0 + (lane & 15)] = (_Float16)acc[e];
    }
  }
}

// ---------------------------------------------------------------------------
// 3x3 conv (pad 1) as implicit-im2col WMMA GEMM.
// A: r1 f16 [12544][512]; W: repacked f16 [512][4608] (k=(kh*3+kw)*512+ci).
// ---------------------------------------------------------------------------
__global__ __launch_bounds__(256) void conv3x3_wmma(
    const _Float16* __restrict__ Ah,
    const _Float16* __restrict__ Wh,
    float* __restrict__ outf)
{
  const int lane = threadIdx.x & 31;
  const int wave = threadIdx.x >> 5;
  const int m0 = blockIdx.x << 4;
  const int n0 = (blockIdx.y << 7) + (wave << 4);
  const int mr = m0 + (lane & 15);
  const int b  = mr / 784, rr = mr - b*784;
  const int h  = rr / 28,  w  = rr - (rr/28)*28;
  const _Float16* brow = Wh + (size_t)(n0 + (lane & 15))*4608 + ((lane>>4)<<3);

  v8f acc = {};
  for (int k0 = 0; k0 < 4608; k0 += 32){
    Frag fa, fb;
    #pragma unroll
    for (int c = 0; c < 2; ++c){
      int kc = k0 + ((lane>>4)<<3) + (c<<4);
      int p  = kc >> 9, ci = kc & 511;
      int kh = p / 3,  kw = p - (p/3)*3;
      int hn = h + kh - 1, wn = w + kw - 1;
      if ((unsigned)hn < 28u && (unsigned)wn < 28u)
        fa.q[c] = *(const uint4*)(Ah + (size_t)((b*28 + hn)*28 + wn)*512 + ci);
      else
        fa.q[c] = zero4();
    }
    fb.q[0] = *(const uint4*)(brow + k0);
    fb.q[1] = *(const uint4*)(brow + k0 + 16);
    acc = __builtin_amdgcn_wmma_f32_16x16x32_f16(false, fa.v, false, fb.v,
                                                 (short)0, acc, false, false);
  }
  const int col   = n0 + (lane & 15);
  const int rbase = m0 + ((lane >> 4) << 3);
  #pragma unroll
  for (int e = 0; e < 8; ++e)
    outf[(size_t)(rbase + e)*512 + col] = acc[e];
}

// ---------------------------------------------------------------------------
// Host launcher
// ---------------------------------------------------------------------------
extern "C" void kernel_launch(void* const* d_in, const int* in_sizes, int n_in,
                              void* d_out, int out_size, void* d_ws, size_t ws_size,
                              hipStream_t stream)
{
  (void)in_sizes; (void)n_in; (void)out_size; (void)ws_size;
  const float* x      = (const float*)d_in[0];
  const float* ln1_w  = (const float*)d_in[1];
  const float* ln1_b  = (const float*)d_in[2];
  const float* qkv_w  = (const float*)d_in[3];
  const float* qkv_b  = (const float*)d_in[4];
  const float* proj_w = (const float*)d_in[5];
  const float* proj_b = (const float*)d_in[6];
  const float* rph    = (const float*)d_in[7];
  const float* rpw    = (const float*)d_in[8];
  const float* ln2_w  = (const float*)d_in[9];
  const float* ln2_b  = (const float*)d_in[10];
  const float* fc1_w  = (const float*)d_in[11];
  const float* fc1_b  = (const float*)d_in[12];
  const float* fc2_w  = (const float*)d_in[13];
  const float* fc2_b  = (const float*)d_in[14];
  const float* c1_w   = (const float*)d_in[15];
  const float* n1_w   = (const float*)d_in[16];
  const float* n1_b   = (const float*)d_in[17];
  const float* c2_w   = (const float*)d_in[18];
  const float* n2_w   = (const float*)d_in[19];
  const float* n2_b   = (const float*)d_in[20];
  const float* c3_w   = (const float*)d_in[21];
  const float* n3_w   = (const float*)d_in[22];
  const float* n3_b   = (const float*)d_in[23];

  size_t off = 0;
  char* base = (char*)d_ws;
  auto carve = [&](size_t bytes) -> char* {
    char* p = base + off;
    off += (bytes + 255) & ~(size_t)255;
    return p;
  };
  // f16 weights
  _Float16* wqkv_h = (_Float16*)carve((size_t)3072*1024*2);
  _Float16* wprj_h = (_Float16*)carve((size_t)1024*1024*2);
  _Float16* wfc1_h = (_Float16*)carve((size_t)4096*1024*2);
  _Float16* wfc2_h = (_Float16*)carve((size_t)1024*4096*2);
  _Float16* wc1_h  = (_Float16*)carve((size_t)512*1024*2);
  _Float16* wc2_h  = (_Float16*)carve((size_t)512*4608*2);
  _Float16* wc3_h  = (_Float16*)carve((size_t)1024*512*2);
  // activations
  _Float16* a_h   = (_Float16*)carve((size_t)TOKENS*1024*2);  // LN1 out (win order)
  _Float16* qkv_h = (_Float16*)carve((size_t)TOKENS*3072*2);  // qkv (win order)
  _Float16* ao_h  = (_Float16*)carve((size_t)TOKENS*1024*2);  // attn out (win order)
  float*    x1_f  = (float*)   carve((size_t)TOKENS*1024*4);  // x + attn (spatial)
  _Float16* a2_h  = (_Float16*)carve((size_t)TOKENS*1024*2);  // LN2 out
  _Float16* h_h   = (_Float16*)carve((size_t)TOKENS*4096*2);  // gelu(fc1)
  float*    x2_f  = (float*)   carve((size_t)TOKENS*1024*4);  // x1 + fc2
  _Float16* x2_h  = (_Float16*)carve((size_t)TOKENS*1024*2);
  float*    r1_f  = (float*)   carve((size_t)TOKENS*512*4);
  _Float16* r1_h  = (_Float16*)carve((size_t)TOKENS*512*2);
  float*    r2_f  = (float*)   carve((size_t)TOKENS*512*4);
  _Float16* r2_h  = (_Float16*)carve((size_t)TOKENS*512*2);
  float*    r3_f  = (float*)   carve((size_t)TOKENS*1024*4);

  // --- weight conversion ---
  cvt_f16<<<1024, 256, 0, stream>>>(qkv_w,  wqkv_h, 3072*1024);
  cvt_f16<<<1024, 256, 0, stream>>>(proj_w, wprj_h, 1024*1024);
  cvt_f16<<<1024, 256, 0, stream>>>(fc1_w,  wfc1_h, 4096*1024);
  cvt_f16<<<1024, 256, 0, stream>>>(fc2_w,  wfc2_h, 1024*4096);
  cvt_f16<<<1024, 256, 0, stream>>>(c1_w,   wc1_h,  512*1024);
  repack_conv2<<<1024, 256, 0, stream>>>(c2_w, wc2_h);
  cvt_f16<<<1024, 256, 0, stream>>>(c3_w,   wc3_h,  1024*512);

  // --- attention branch ---
  ln_rows<<<TOKENS, 256, 0, stream>>>(x, ln1_w, ln1_b, a_h, 1024, 1e-5f, 1, 0);
  gemm_wmma<EPI_BIAS_F16><<<dim3(784,24), 256, 16*1024*2, stream>>>(
      a_h, wqkv_h, qkv_b, nullptr, nullptr, qkv_h, TOKENS, 3072, 1024);
  attn_win<<<1024, 256, 0, stream>>>(qkv_h, rph, rpw, ao_h);
  gemm_wmma<EPI_BIAS_RES_REMAP_F32><<<dim3(784,8), 256, 16*1024*2, stream>>>(
      ao_h, wprj_h, proj_b, x, x1_f, nullptr, TOKENS, 1024, 1024);

  // --- MLP branch ---
  ln_rows<<<TOKENS, 256, 0, stream>>>(x1_f, ln2_w, ln2_b, a2_h, 1024, 1e-5f, 0, 0);
  gemm_wmma<EPI_BIAS_GELU_F16><<<dim3(784,32), 256, 16*1024*2, stream>>>(
      a2_h, wfc1_h, fc1_b, nullptr, nullptr, h_h, TOKENS, 4096, 1024);
  gemm_wmma<EPI_BIAS_RES_F32_F16><<<dim3(784,8), 256, 16*4096*2, stream>>>(
      h_h, wfc2_h, fc2_b, x1_f, x2_f, x2_h, TOKENS, 1024, 4096);

  // --- ResBottleneckBlock ---
  gemm_wmma<EPI_F32><<<dim3(784,4), 256, 16*1024*2, stream>>>(
      x2_h, wc1_h, nullptr, nullptr, r1_f, nullptr, TOKENS, 512, 1024);
  ln_rows<<<TOKENS, 256, 0, stream>>>(r1_f, n1_w, n1_b, r1_h, 512, 1e-6f, 0, 1);
  conv3x3_wmma<<<dim3(784,4), 256, 0, stream>>>(r1_h, wc2_h, r2_f);
  ln_rows<<<TOKENS, 256, 0, stream>>>(r2_f, n2_w, n2_b, r2_h, 512, 1e-6f, 0, 1);
  gemm_wmma<EPI_F32><<<dim3(784,8), 256, 16*512*2, stream>>>(
      r2_h, wc3_h, nullptr, nullptr, r3_f, nullptr, TOKENS, 1024, 512);
  final_ln_add<<<TOKENS, 256, 0, stream>>>(r3_f, n3_w, n3_b, x2_f, (float*)d_out);
}
